// QuantizedLinear4bit_31868657336598
// MI455X (gfx1250) — 8-bit matrix operand order repaired and checked by reading, compiled, not run on hardware
//
#include <hip/hip_runtime.h>
#include <hip/hip_bf16.h>
#include <stdint.h>


typedef __attribute__((ext_vector_type(8))) int v8i;

#define DIN   4096
#define DOUT  4096
#define MROWS 8192
#define QMAXF 7.0f

__global__ void ql_init_kernel(unsigned* wsmax) {
    if (threadIdx.x == 0) wsmax[0] = 0u;
}

__global__ void ql_absmax_kernel(const float* __restrict__ x,
                                 unsigned* __restrict__ wsmax, int n4) {
    int idx    = blockIdx.x * blockDim.x + threadIdx.x;
    int stride = gridDim.x * blockDim.x;
    const float4* x4 = (const float4*)x;
    float m = 0.0f;
    for (int i = idx; i < n4; i += stride) {
        float4 v = x4[i];
        m = fmaxf(m, fmaxf(fmaxf(fabsf(v.x), fabsf(v.y)),
                           fmaxf(fabsf(v.z), fabsf(v.w))));
    }
    for (int off = 16; off > 0; off >>= 1)
        m = fmaxf(m, __shfl_xor(m, off, 32));
    if ((threadIdx.x & 31) == 0)
        atomicMax(wsmax, __float_as_uint(m));
}

__global__ void ql_quant_kernel(const float* __restrict__ x,
                                signed char* __restrict__ xq,
                                const unsigned* __restrict__ wsmax, int n8) {
    int i = blockIdx.x * blockDim.x + threadIdx.x;
    if (i >= n8) return;
    const float mx    = __uint_as_float(*wsmax);
    const float scale = fmaxf(mx * (1.0f / QMAXF), 1e-8f);

    const float4* x4 = (const float4*)x;
    float4 a = x4[2 * i + 0];
    float4 b = x4[2 * i + 1];

    int q[8];
    float v[8] = {a.x, a.y, a.z, a.w, b.x, b.y, b.z, b.w};
#pragma unroll
    for (int j = 0; j < 8; ++j) {
        float r = rintf(v[j] / scale);
        r = fminf(fmaxf(r, -8.0f), 7.0f);
        q[j] = (int)r;
    }
    int2 packed;
    packed.x = (q[0] & 0xff) | ((q[1] & 0xff) << 8) |
               ((q[2] & 0xff) << 16) | ((q[3] & 0xff) << 24);
    packed.y = (q[4] & 0xff) | ((q[5] & 0xff) << 8) |
               ((q[6] & 0xff) << 16) | ((q[7] & 0xff) << 24);
    ((int2*)xq)[i] = packed;
}

__global__ __launch_bounds__(128) void ql_gemm_kernel(
    const signed char* __restrict__ xq,
    const signed char* __restrict__ qw,
    const float* __restrict__ bias,
    const unsigned* __restrict__ wsmax,
    const float* __restrict__ wscale,
    float* __restrict__ out)
{
    const int lane  = threadIdx.x & 31;
    const int wave  = threadIdx.x >> 5;
    const int mlane = lane & 15;
    const int half  = lane >> 4;

    const int waveM = blockIdx.x * 128 + (wave & 1) * 64;
    const int waveN = blockIdx.y * 128 + (wave >> 1) * 64;

    v8i c[4][4] = {};

    const signed char* aBase[4];
    const signed char* bBase[4];
#pragma unroll
    for (int t = 0; t < 4; ++t) {
        aBase[t] = xq + (size_t)(waveM + t * 16 + mlane) * DIN + half * 8;
        bBase[t] = qw + (size_t)(waveN + t * 16 + mlane) * DIN + half * 8;
    }

#pragma unroll 1
    for (int k = 0; k < DIN; k += 64) {
        v8i a[4], b[4];
#pragma unroll
        for (int t = 0; t < 4; ++t) {
            const signed char* p = aBase[t] + k;
            int2 t0 = *(const int2*)(p + 0);
            int2 t1 = *(const int2*)(p + 16);
            int2 t2 = *(const int2*)(p + 32);
            int2 t3 = *(const int2*)(p + 48);
            a[t][0] = t0.x; a[t][1] = t0.y;
            a[t][2] = t1.x; a[t][3] = t1.y;
            a[t][4] = t2.x; a[t][5] = t2.y;
            a[t][6] = t3.x; a[t][7] = t3.y;
        }
#pragma unroll
        for (int t = 0; t < 4; ++t) {
            const signed char* p = bBase[t] + k;
            int2 u0 = *(const int2*)(p + 0);
            int2 u1 = *(const int2*)(p + 16);
            int2 u2 = *(const int2*)(p + 32);
            int2 u3 = *(const int2*)(p + 48);
            b[t][0] = u0.x; b[t][1] = u0.y; b[t][2] = u1.x; b[t][3] = u1.y;
            b[t][4] = u2.x; b[t][5] = u2.y; b[t][6] = u3.x; b[t][7] = u3.y;
        }
#pragma unroll
        for (int tm = 0; tm < 4; ++tm)
#pragma unroll
            for (int tn = 0; tn < 4; ++tn)
                c[tm][tn] = __builtin_amdgcn_wmma_i32_16x16x64_iu8(
                     true, a[tm],  true, b[tn],
                    c[tm][tn],  false,  false);
    }

    const float s = fmaxf(__uint_as_float(*wsmax) * (1.0f / QMAXF), 1e-8f)
                    * wscale[0];

#pragma unroll
    for (int tn = 0; tn < 4; ++tn) {
        const int n  = waveN + tn * 16 + mlane;
        const float bv = bias[n];
#pragma unroll
        for (int tm = 0; tm < 4; ++tm) {
            const int mbase = waveM + tm * 16 + half * 8;
#pragma unroll
            for (int r = 0; r < 8; ++r) {
                out[(size_t)(mbase + r) * DOUT + n] =
                    (float)c[tm][tn][r] * s + bv;
            }
        }
    }
}

extern "C" void kernel_launch(void* const* d_in, const int* in_sizes, int n_in,
                              void* d_out, int out_size, void* d_ws, size_t ws_size,
                              hipStream_t stream) {
    const float*       x      = (const float*)d_in[0];
    const signed char* qw     = (const signed char*)d_in[1];
    const float*       wscale = (const float*)d_in[2];
    const float*       bias   = (const float*)d_in[3];
    float*             out    = (float*)d_out;

    unsigned*    wsmax = (unsigned*)d_ws;
    signed char* xq    = (signed char*)d_ws + 256;

    const int nElem = MROWS * DIN;

    ql_init_kernel<<<1, 64, 0, stream>>>(wsmax);
    ql_absmax_kernel<<<2048, 256, 0, stream>>>(x, wsmax, nElem / 4);
    ql_quant_kernel<<<(nElem / 8 + 255) / 256, 256, 0, stream>>>(x, xq, wsmax,
                                                                 nElem / 8);
    dim3 grid(MROWS / 128, DOUT / 128);
    ql_gemm_kernel<<<grid, 128, 0, stream>>>(xq, qw, bias, wsmax, wscale, out);
}
